// RPN_25967372272003
// MI455X (gfx1250) — matrix operand order repaired and checked by reading, not compiled, not run on hardware
//
#include <hip/hip_runtime.h>

// ---------------------------------------------------------------------------
// Types for CDNA5 WMMA (bf16 in, f32 accum)
// ---------------------------------------------------------------------------
typedef __bf16 bf16_t;
typedef __attribute__((ext_vector_type(16))) __bf16 v16bf;
typedef __attribute__((ext_vector_type(8)))  __bf16 v8bf;
typedef __attribute__((ext_vector_type(8)))  float  v8f;

#define NEG_SCORE (-1e30f)
#define NEG_HALF  (-5e29f)
#define NMS_THR   0.7f
#define MAX_OUT   2000
#define NA        82944      // 96*96*9 anchors
#define GP        9216       // 96*96 pixels
#define CIN       256
#define WPAD      98         // 96 + 1-pixel halo each side

__device__ __forceinline__ bf16_t f2bf(float f) {
    union { float f; unsigned u; } a; a.f = f;
    unsigned r = a.u + 0x7FFFu + ((a.u >> 16) & 1u);   // round-nearest-even
    unsigned short h = (unsigned short)(r >> 16);
    return __builtin_bit_cast(__bf16, h);
}

__device__ __forceinline__ v16bf load_a_frag(const bf16_t* base) {
    v8bf a0 = *(const v8bf*)(base);
    v8bf a1 = *(const v8bf*)(base + 8);
    return __builtin_shufflevector(a0, a1,
            0,1,2,3,4,5,6,7,8,9,10,11,12,13,14,15);
}

__device__ __forceinline__ v8f wmma_bf16(v16bf a, v16bf b, v8f c) {
    return __builtin_amdgcn_wmma_f32_16x16x32_bf16(
               false, a, false, b, (short)0, c, false, false);
}

// ---------------------------------------------------------------------------
// Prep kernels
// ---------------------------------------------------------------------------
// Wp2[t][o][i] = conv_w[o][i][ky][kx],  t = ky*3+kx   (OIHW flat: o*2304 + i*9 + t)
__global__ void pack_w1(const float* __restrict__ conv_w, bf16_t* __restrict__ Wp2) {
    int idx = blockIdx.x * 256 + threadIdx.x;        // 9*256*256
    if (idx >= 9 * 256 * 256) return;
    int t = idx >> 16;
    int r = idx & 65535;
    int o = r >> 8;
    int i = r & 255;
    Wp2[idx] = f2bf(conv_w[(size_t)o * 2304 + (size_t)i * 9 + t]);
}

// Wsmall[row][c]: rows 0..35 bbox_w, 36..53 score_w, 54..63 zero
__global__ void pack_w2(const float* __restrict__ bbox_w, const float* __restrict__ score_w,
                        bf16_t* __restrict__ Ws) {
    int idx = blockIdx.x * 256 + threadIdx.x;        // 64*256
    if (idx >= 64 * 256) return;
    int row = idx >> 8, c = idx & 255;
    float v = 0.0f;
    if (row < 36) v = bbox_w[row * 256 + c];
    else if (row < 54) v = score_w[(row - 36) * 256 + c];
    Ws[idx] = f2bf(v);
}

// zero the padded activation buffer (as 32-bit stores; element count is even)
__global__ void fill_zero(unsigned* __restrict__ p, int n32) {
    int idx = blockIdx.x * 256 + threadIdx.x;
    if (idx < n32) p[idx] = 0u;
}

// Xtp[(y+1)*98 + (x+1)][c] = x[c][y*96+x]  (bf16, halo stays zero)
__global__ void make_xt(const float* __restrict__ x, bf16_t* __restrict__ Xtp) {
    int idx = blockIdx.x * 256 + threadIdx.x;        // 256*9216
    if (idx >= CIN * GP) return;
    int c = idx / GP;
    int p = idx - c * GP;
    int y = p / 96;
    int xx = p - y * 96;
    Xtp[(size_t)((y + 1) * WPAD + (xx + 1)) * CIN + c] = f2bf(x[idx]);
}

// ---------------------------------------------------------------------------
// Conv 3x3 SAME + bias + ReLU as 9 shifted WMMA GEMMs over the padded buffer.
// Each wave computes a 32x64 tile (2 channel-subtiles x 4 pixel-subtiles):
// 8 accumulators, 2 A-frags + 4 B-frags per k-step -> 8 WMMAs per k-step,
// 1.5 loads/WMMA, branch-free inner loop with immediate-offset addressing.
// Output transposed: Ht[p][c] (bf16) for GEMM2.
// ---------------------------------------------------------------------------
__global__ void __launch_bounds__(256)
conv3x3_wmma(const bf16_t* __restrict__ Wp2, const bf16_t* __restrict__ Xtp,
             const float* __restrict__ conv_b, bf16_t* __restrict__ Ht) {
    const int lane   = threadIdx.x & 31;
    const int wv     = threadIdx.x >> 5;
    const int tile   = blockIdx.x * 8 + wv;          // 1152 tiles total
    const int mtile  = tile & 7;                     // 8 tiles of 32 channels
    const int ntile  = tile >> 3;                    // 144 tiles of 64 pixels
    const int lane16 = lane & 15;
    const int hi     = lane >> 4;                    // lane half
    const int p_base = ntile * 64;
    const int m0     = mtile * 32 + lane16;          // A row, subtile 0
    const int koffA  = hi * 16;                      // A: K {0..15} | {16..31}
    const int koffB  = hi * 16;                      // B: K {0..15} | {16..31}

    // Per pixel-subtile base pointers (each 16-pixel group stays in one row).
    const bf16_t* Pb[4];
#pragma unroll
    for (int s = 0; s < 4; ++s) {
        const int ps = p_base + s * 16;
        const int ys = ps / 96;
        const int xs = ps - ys * 96;
        Pb[s] = Xtp + (size_t)(ys * WPAD + xs + lane16) * CIN + koffB;
    }
    const bf16_t* Ab = Wp2 + (size_t)m0 * 256 + koffA;

    v8f acc[2][4] = {};
#pragma unroll
    for (int t = 0; t < 9; ++t) {
        const int ky = t / 3, kx = t % 3;
        const size_t aoff = (size_t)t * 65536;
        const size_t boff = (size_t)(ky * WPAD + kx) * CIN;   // folds to immediates
#pragma unroll
        for (int k0 = 0; k0 < 256; k0 += 32) {
            v16bf a0 = load_a_frag(Ab + aoff + k0);
            v16bf a1 = load_a_frag(Ab + aoff + 16 * 256 + k0);
            v16bf b0 = *(const v16bf*)(Pb[0] + boff + k0);
            v16bf b1 = *(const v16bf*)(Pb[1] + boff + k0);
            v16bf b2 = *(const v16bf*)(Pb[2] + boff + k0);
            v16bf b3 = *(const v16bf*)(Pb[3] + boff + k0);
            acc[0][0] = wmma_bf16(a0, b0, acc[0][0]);
            acc[0][1] = wmma_bf16(a0, b1, acc[0][1]);
            acc[0][2] = wmma_bf16(a0, b2, acc[0][2]);
            acc[0][3] = wmma_bf16(a0, b3, acc[0][3]);
            acc[1][0] = wmma_bf16(a1, b0, acc[1][0]);
            acc[1][1] = wmma_bf16(a1, b1, acc[1][1]);
            acc[1][2] = wmma_bf16(a1, b2, acc[1][2]);
            acc[1][3] = wmma_bf16(a1, b3, acc[1][3]);
        }
    }
    // Epilogue: bias + ReLU, store transposed bf16.
    const int pb = p_base + lane16;                  // D: N = lane%16
    const int cb = mtile * 32 + hi * 8;              // D: M = v + hi*8 (+msub*16)
    bf16_t* Hb = Ht + (size_t)pb * CIN + cb;
#pragma unroll
    for (int ms = 0; ms < 2; ++ms) {
        const int cofs = ms * 16;
#pragma unroll
        for (int v = 0; v < 8; ++v) {
            const float bias = conv_b[cb + cofs + v];
#pragma unroll
            for (int s = 0; s < 4; ++s) {
                Hb[(size_t)s * 16 * CIN + cofs + v] =
                    f2bf(fmaxf(acc[ms][s][v] + bias, 0.0f));
            }
        }
    }
}

// ---------------------------------------------------------------------------
// Fused 1x1 convs (bbox 36 + score 18, padded to 64 rows) as one WMMA GEMM.
// bregraw[c][p] (36x9216), objraw[c][p] (18x9216), bias + ReLU applied.
// ---------------------------------------------------------------------------
__global__ void __launch_bounds__(256)
gemm1x1_wmma(const bf16_t* __restrict__ Ws, const bf16_t* __restrict__ Ht,
             const float* __restrict__ bbox_b, const float* __restrict__ score_b,
             float* __restrict__ bregraw, float* __restrict__ objraw) {
    const int lane   = threadIdx.x & 31;
    const int wv     = threadIdx.x >> 5;
    const int tile   = blockIdx.x * 8 + wv;          // 2304 tiles (4 x 576)
    const int mtile  = tile & 3;
    const int ntile  = tile >> 2;
    const int lane16 = lane & 15;
    const int hi     = lane >> 4;
    const int m      = mtile * 16 + lane16;
    const int p      = ntile * 16 + lane16;
    const int koffA  = hi * 16;
    const int koffB  = hi * 16;
    const bf16_t* Ab = Ws + (size_t)m * 256 + koffA;
    const bf16_t* Bb = Ht + (size_t)p * CIN + koffB;

    v8f acc = {};
#pragma unroll
    for (int k0 = 0; k0 < 256; k0 += 32) {
        v16bf af  = load_a_frag(Ab + k0);
        v16bf bfv = *(const v16bf*)(Bb + k0);
        acc = wmma_bf16(af, bfv, acc);
    }
#pragma unroll
    for (int v = 0; v < 8; ++v) {
        const int row = mtile * 16 + hi * 8 + v;
        if (row < 36) {
            float val = fmaxf(acc[v] + bbox_b[row], 0.0f);
            bregraw[(size_t)row * GP + p] = val;
        } else if (row < 54) {
            float val = fmaxf(acc[v] + score_b[row - 36], 0.0f);
            objraw[(size_t)(row - 36) * GP + p] = val;
        }
    }
}

// ---------------------------------------------------------------------------
// Anchor decode: boxes + gated scores.
// breg[a][k] = bregraw_flat[4a+k]; obj[a][k] = objraw_flat[2a+k].
// Anchor a: t=a%9, j=(a/9)%96, i=(a/9)/96; cxa=(i+.5)*16 (X varies along axis0!)
// ---------------------------------------------------------------------------
__global__ void decode_kernel(const float* __restrict__ bregraw,
                              const float* __restrict__ objraw,
                              float* __restrict__ boxes, float* __restrict__ sarr) {
    int a = blockIdx.x * 256 + threadIdx.x;
    if (a >= NA) return;
    int t = a % 9;
    int rem = a / 9;
    int jj = rem % 96;
    int ii = rem / 96;
    float cxa = (ii + 0.5f) * 16.0f;
    float cya = (jj + 0.5f) * 16.0f;
    const float areas[3] = {16384.0f, 65536.0f, 262144.0f};
    int ai = t / 3, ri = t % 3;
    float wr = (ri == 1) ? 2.0f : 1.0f;
    float hr = (ri == 2) ? 2.0f : 1.0f;
    float u  = sqrtf(areas[ai] / (wr * hr));
    float wa = wr * u, ha = hr * u;

    float b0 = bregraw[4 * a + 0];
    float b1 = bregraw[4 * a + 1];
    float b2 = bregraw[4 * a + 2];
    float b3 = bregraw[4 * a + 3];
    float cx = b0 * wa + cxa;
    float cy = b1 * ha + cya;
    float w  = expf(b2) * wa;
    float h  = expf(b3) * ha;
    float x1 = fminf(fmaxf(cx - 0.5f * w, 0.0f), 1536.0f);
    float y1 = fminf(fmaxf(cy - 0.5f * h, 0.0f), 1536.0f);
    float x2 = fminf(fmaxf(cx + 0.5f * w, 0.0f), 1536.0f);
    float y2 = fminf(fmaxf(cy + 0.5f * h, 0.0f), 1536.0f);
    boxes[4 * a + 0] = x1; boxes[4 * a + 1] = y1;
    boxes[4 * a + 2] = x2; boxes[4 * a + 3] = y2;

    float score0 = objraw[2 * a];
    float area   = fabsf((x1 - x2) * (y1 - y2));
    sarr[a] = (score0 > 0.2f && area > 100.0f) ? score0 : NEG_SCORE;
}

// ---------------------------------------------------------------------------
// Persistent single-block greedy NMS. Deterministic LDS-scan compaction,
// then 2000 serial argmax+suppress steps with early exit.
// out layout: [2000*4 boxes][2000*2 scores]
// ---------------------------------------------------------------------------
__global__ void __launch_bounds__(1024)
nms_kernel(const float* __restrict__ boxes, const float* __restrict__ sarr,
           const float* __restrict__ objraw,
           float* __restrict__ cbox, float* __restrict__ cs, int* __restrict__ cidx,
           float* __restrict__ out) {
    __shared__ float sv[1024];
    __shared__ int   si[1024];
    __shared__ int   s_n;
    __shared__ float s_bj[4];
    __shared__ int   s_ok;
    const int tid = threadIdx.x;

    // ---- deterministic compaction (82944 = 81 chunks of 1024) ----
    if (tid == 0) s_n = 0;
    __syncthreads();
    for (int base = 0; base < NA; base += 1024) {
        int a = base + tid;
        float v = sarr[a];
        int keep = (v > NEG_HALF) ? 1 : 0;
        si[tid] = keep;
        __syncthreads();
        for (int off = 1; off < 1024; off <<= 1) {     // inclusive scan
            int add = (tid >= off) ? si[tid - off] : 0;
            __syncthreads();
            si[tid] += add;
            __syncthreads();
        }
        int pos0 = s_n;
        if (keep) {
            int pos = pos0 + si[tid] - 1;
            cidx[pos] = a;
            cs[pos]   = v;
            cbox[pos * 4 + 0] = boxes[a * 4 + 0];
            cbox[pos * 4 + 1] = boxes[a * 4 + 1];
            cbox[pos * 4 + 2] = boxes[a * 4 + 2];
            cbox[pos * 4 + 3] = boxes[a * 4 + 3];
        }
        __syncthreads();
        if (tid == 0) s_n += si[1023];
        __syncthreads();
    }
    const int n = s_n;

    // ---- serial greedy loop ----
    int it = 0;
    for (; it < MAX_OUT; ++it) {
        float bv = -3e38f; int bi = -1;
        for (int i = tid; i < n; i += 1024) {
            float v = cs[i];
            if (v > bv) { bv = v; bi = i; }
        }
        sv[tid] = bv; si[tid] = bi;
        __syncthreads();
        for (int off = 512; off > 0; off >>= 1) {
            if (tid < off) {
                float v2 = sv[tid + off]; int i2 = si[tid + off];
                if (v2 > sv[tid] || (v2 == sv[tid] && i2 >= 0 && i2 < si[tid])) {
                    sv[tid] = v2; si[tid] = i2;
                }
            }
            __syncthreads();
        }
        if (tid == 0) {
            int j = si[0];
            int ok = (j >= 0) && (sv[0] > NEG_HALF);
            s_ok = ok;
            if (ok) {
                float b0 = cbox[j * 4 + 0], b1 = cbox[j * 4 + 1];
                float b2 = cbox[j * 4 + 2], b3 = cbox[j * 4 + 3];
                s_bj[0] = b0; s_bj[1] = b1; s_bj[2] = b2; s_bj[3] = b3;
                out[it * 4 + 0] = b0; out[it * 4 + 1] = b1;
                out[it * 4 + 2] = b2; out[it * 4 + 3] = b3;
                int a = cidx[j];
                out[MAX_OUT * 4 + it * 2 + 0] = objraw[2 * a];
                out[MAX_OUT * 4 + it * 2 + 1] = objraw[2 * a + 1];
                cs[j] = NEG_SCORE;
            }
        }
        __syncthreads();
        if (!s_ok) break;
        float b0 = s_bj[0], b1 = s_bj[1], b2 = s_bj[2], b3 = s_bj[3];
        float a1 = (b2 - b0) * (b3 - b1);
        for (int i = tid; i < n; i += 1024) {
            float x1 = fmaxf(b0, cbox[i * 4 + 0]);
            float y1 = fmaxf(b1, cbox[i * 4 + 1]);
            float x2 = fminf(b2, cbox[i * 4 + 2]);
            float y2 = fminf(b3, cbox[i * 4 + 3]);
            float inter = fmaxf(x2 - x1, 0.0f) * fmaxf(y2 - y1, 0.0f);
            float a2 = (cbox[i * 4 + 2] - cbox[i * 4 + 0]) *
                       (cbox[i * 4 + 3] - cbox[i * 4 + 1]);
            float iou = inter / (a1 + a2 - inter + 1e-9f);
            if (iou > NMS_THR) cs[i] = NEG_SCORE;
        }
        __syncthreads();
    }
    // ---- zero-fill remaining (invalid) rows ----
    for (int k = it * 4 + tid; k < MAX_OUT * 4; k += 1024) out[k] = 0.0f;
    for (int k = MAX_OUT * 4 + it * 2 + tid; k < MAX_OUT * 6; k += 1024) out[k] = 0.0f;
}

// ---------------------------------------------------------------------------
// Host launcher
// ---------------------------------------------------------------------------
extern "C" void kernel_launch(void* const* d_in, const int* in_sizes, int n_in,
                              void* d_out, int out_size, void* d_ws, size_t ws_size,
                              hipStream_t stream) {
    const float* x       = (const float*)d_in[0];   // 1x256x96x96
    const float* conv_w  = (const float*)d_in[1];   // 256x256x3x3
    const float* conv_b  = (const float*)d_in[2];   // 256
    const float* bbox_w  = (const float*)d_in[3];   // 36x256
    const float* bbox_b  = (const float*)d_in[4];   // 36
    const float* score_w = (const float*)d_in[5];   // 18x256
    const float* score_b = (const float*)d_in[6];   // 18
    float* out = (float*)d_out;                     // 2000*4 + 2000*2

    char* ws = (char*)d_ws;
    size_t off = 0;
    auto take = [&](size_t bytes) { char* p = ws + off; off = (off + bytes + 255) & ~(size_t)255; return p; };
    bf16_t* Wp2     = (bf16_t*)take((size_t)9 * 256 * 256 * 2);
    bf16_t* Ws      = (bf16_t*)take((size_t)64 * 256 * 2);
    bf16_t* Xtp     = (bf16_t*)take((size_t)WPAD * WPAD * CIN * 2);   // padded
    bf16_t* Ht      = (bf16_t*)take((size_t)GP * CIN * 2);
    float*  bregraw = (float*)take((size_t)36 * GP * 4);
    float*  objraw  = (float*)take((size_t)18 * GP * 4);
    float*  boxes   = (float*)take((size_t)NA * 4 * 4);
    float*  sarr    = (float*)take((size_t)NA * 4);
    float*  cbox    = (float*)take((size_t)NA * 4 * 4);
    float*  cs      = (float*)take((size_t)NA * 4);
    int*    cidx    = (int*)take((size_t)NA * 4);
    (void)ws_size; (void)in_sizes; (void)n_in; (void)out_size;

    const int nfill32 = (WPAD * WPAD * CIN) / 2;     // bf16 pairs
    pack_w1<<<(9 * 256 * 256 + 255) / 256, 256, 0, stream>>>(conv_w, Wp2);
    pack_w2<<<(64 * 256 + 255) / 256, 256, 0, stream>>>(bbox_w, score_w, Ws);
    fill_zero<<<(nfill32 + 255) / 256, 256, 0, stream>>>((unsigned*)Xtp, nfill32);
    make_xt<<<(CIN * GP + 255) / 256, 256, 0, stream>>>(x, Xtp);

    conv3x3_wmma<<<(8 * 144) / 8, 256, 0, stream>>>(Wp2, Xtp, conv_b, Ht);
    gemm1x1_wmma<<<(4 * 576) / 8, 256, 0, stream>>>(Ws, Ht, bbox_b, score_b, bregraw, objraw);

    decode_kernel<<<NA / 256, 256, 0, stream>>>(bregraw, objraw, boxes, sarr);
    nms_kernel<<<1, 1024, 0, stream>>>(boxes, sarr, objraw, cbox, cs, cidx, out);
}
